// DGCNN_45921790329152
// MI455X (gfx1250) — matrix operand order repaired and checked by reading, not compiled, not run on hardware
//
#include <hip/hip_runtime.h>

#define NUM_GRAPHS 1000
#define NODES_PER  100
#define EDGES_PER  1600
#define HIDDEN     128
#define NFEAT      128
#define KTOP       40
#define LATENT     385
#define NNODES     (NUM_GRAPHS * NODES_PER)   // 100000
#define NEDGES     (NUM_GRAPHS * EDGES_PER)   // 1600000

typedef __bf16 bf16;
typedef __attribute__((ext_vector_type(16))) __bf16 v16bf;
typedef __attribute__((ext_vector_type(8)))  float  v8f;

// ---------------------------------------------------------------- utilities
__global__ void k_zero_i32(int* p, int n) {
    int i = blockIdx.x * blockDim.x + threadIdx.x;
    if (i < n) p[i] = 0;
}

__global__ void k_deg(const int* __restrict__ ei, int* __restrict__ deg, int E) {
    int e = blockIdx.x * blockDim.x + threadIdx.x;
    if (e < E) atomicAdd(&deg[ei[E + e]], 1);   // dst row of edge_index
}

__global__ void k_dinv(const int* __restrict__ deg, float* __restrict__ dinv, int n) {
    int i = blockIdx.x * blockDim.x + threadIdx.x;
    if (i < n) {
        int d = deg[i];
        dinv[i] = (d > 0) ? rsqrtf((float)d) : 0.0f;
    }
}

__global__ void k_norm(const int* __restrict__ ei, const float* __restrict__ dinv,
                       float* __restrict__ norm, int E) {
    int e = blockIdx.x * blockDim.x + threadIdx.x;
    if (e < E) norm[e] = dinv[ei[e]] * dinv[ei[E + e]];
}

// h0[n, 0:128] = z_table[z[n]], h0[n, 128:256] = x[n]   (bf16)
__global__ void k_embed(const int* __restrict__ z, const float* __restrict__ z_table,
                        const float* __restrict__ x, bf16* __restrict__ h0) {
    long idx = (long)blockIdx.x * blockDim.x + threadIdx.x;
    if (idx >= (long)NNODES * 256) return;
    int n = (int)(idx >> 8);
    int c = (int)(idx & 255);
    float v = (c < 128) ? z_table[(long)z[n] * 128 + c] : x[(long)n * 128 + (c - 128)];
    h0[idx] = (bf16)v;
}

// Wt[o, k] = W[k, o]  (f32 -> bf16 transpose; W is [fin, fout])
__global__ void k_cvtT(const float* __restrict__ W, bf16* __restrict__ Wt, int fin, int fout) {
    int idx = blockIdx.x * blockDim.x + threadIdx.x;
    if (idx >= fin * fout) return;
    int o = idx / fin, k = idx % fin;
    Wt[idx] = (bf16)W[(long)k * fout + o];
}

// ------------------------------------------------- GEMM: C[N,fout] = A[N,K] @ Wt[fout,K]^T
// one wave computes one 16x16 C tile via v_wmma_f32_16x16x32_bf16
__global__ void k_gemm_wmma(const bf16* __restrict__ A, const bf16* __restrict__ Bt,
                            float* __restrict__ C, int N, int K, int fout) {
    int wave = blockIdx.x * (blockDim.x >> 5) + (threadIdx.x >> 5);
    int lane = threadIdx.x & 31;
    int tiles_n = fout >> 4;
    int total = (N >> 4) * tiles_n;
    if (wave >= total) return;
    int mt = wave / tiles_n;
    int nt = wave % tiles_n;

    int lrow = lane & 15;
    int half = lane >> 4;

    v8f acc = {};
    const bf16* arow = A  + (size_t)(mt * 16 + lrow) * K;
    const bf16* brow = Bt + (size_t)(nt * 16 + lrow) * K;

    for (int k0 = 0; k0 < K; k0 += 32) {
        v16bf a, b;
        // A 16x32 bf16 layout (ISA 7.12.2): half h covers K = 8h..8h+7 then 16+8h..16+8h+7
#pragma unroll
        for (int j = 0; j < 4; ++j) {
            int kk = k0 + 8 * half + 2 * j;
            a[2 * j]     = arow[kk];
            a[2 * j + 1] = arow[kk + 1];
        }
#pragma unroll
        for (int j = 4; j < 8; ++j) {
            int kk = k0 + 16 + 8 * half + 2 * (j - 4);
            a[2 * j]     = arow[kk];
            a[2 * j + 1] = arow[kk + 1];
        }
        // B 32x16 bf16: half h covers K = 8h..8h+7 then 16+8h..16+8h+7 (same as A)
#pragma unroll
        for (int i = 0; i < 16; ++i) b[i] = brow[k0 + 8 * half + i + (i >= 8 ? 8 : 0)];

        acc = __builtin_amdgcn_wmma_f32_16x16x32_bf16(
            false, a, false, b, (short)0, acc, false, false);
    }

    int col = nt * 16 + lrow;
#pragma unroll
    for (int v = 0; v < 8; ++v) {
        int row = mt * 16 + half * 8 + v;     // D: VGPR v -> M=v (lanes 0-15), M=v+8 (16-31)
        C[(size_t)row * fout + col] = acc[v];
    }
}

// ------------------------------------------------- per-graph GCN scatter (128 channels)
// one workgroup per graph; LDS f32 accumulator; ds_add_f32 atomics
__global__ void k_scatter128(const float* __restrict__ hW, const int* __restrict__ ei,
                             const float* __restrict__ norm, const float* __restrict__ bias,
                             bf16* __restrict__ feat, int colOff, bf16* __restrict__ hnext) {
    __shared__ float sacc[NODES_PER * 128];   // 51.2 KB
    int g   = blockIdx.x;
    int tid = threadIdx.x;
    for (int i = tid; i < NODES_PER * 128; i += blockDim.x) sacc[i] = 0.0f;
    __syncthreads();

    int wid  = tid >> 5;
    int lane = tid & 31;
    int nwv  = blockDim.x >> 5;
    int ebase = g * EDGES_PER;
    for (int e = wid; e < EDGES_PER; e += nwv) {
        int eg = ebase + e;
        int s  = ei[eg];
        int d  = ei[NEDGES + eg];
        float nv = norm[eg];
        int dl = (d - g * NODES_PER) * 128;
        const float* row = hW + (size_t)s * 128;
#pragma unroll
        for (int c = lane; c < 128; c += 32)
            atomicAdd(&sacc[dl + c], nv * row[c]);
    }
    __syncthreads();

    for (int i = tid; i < NODES_PER * 128; i += blockDim.x) {
        int node = i >> 7, c = i & 127;
        float v = tanhf(sacc[i] + bias[c]);
        size_t gn = (size_t)g * NODES_PER + node;
        feat[gn * LATENT + colOff + c] = (bf16)v;
        hnext[gn * 128 + c] = (bf16)v;
    }
}

// ------------------------------------------------- layer 3: matvec + scalar scatter
__global__ void k_matvec_w3(const bf16* __restrict__ h, const float* __restrict__ W3,
                            float* __restrict__ hw3, int n) {
    int i = blockIdx.x * blockDim.x + threadIdx.x;
    if (i >= n) return;
    const bf16* row = h + (size_t)i * 128;
    float s = 0.0f;
#pragma unroll
    for (int c = 0; c < 128; ++c) s += (float)row[c] * W3[c];
    hw3[i] = s;
}

__global__ void k_scatter1(const float* __restrict__ hw3, const int* __restrict__ ei,
                           const float* __restrict__ norm, const float* __restrict__ b3,
                           bf16* __restrict__ feat, float* __restrict__ keys) {
    __shared__ float sacc[NODES_PER];
    int g   = blockIdx.x;
    int tid = threadIdx.x;
    if (tid < NODES_PER) sacc[tid] = 0.0f;
    __syncthreads();
    int ebase = g * EDGES_PER;
    for (int e = tid; e < EDGES_PER; e += blockDim.x) {
        int eg = ebase + e;
        atomicAdd(&sacc[ei[NEDGES + eg] - g * NODES_PER], norm[eg] * hw3[ei[eg]]);
    }
    __syncthreads();
    if (tid < NODES_PER) {
        float v = tanhf(sacc[tid] + b3[0]);
        size_t gn = (size_t)g * NODES_PER + tid;
        feat[gn * LATENT + (LATENT - 1)] = (bf16)v;
        keys[gn] = v;
    }
}

// ------------------------------------------------- sort-pool: stable descending rank, top-K
__global__ void k_sortpool(const float* __restrict__ keys, const bf16* __restrict__ feat,
                           bf16* __restrict__ top) {
    __shared__ float skey[NODES_PER];
    __shared__ int   srank[NODES_PER];
    int g   = blockIdx.x;
    int tid = threadIdx.x;
    if (tid < NODES_PER) skey[tid] = keys[(size_t)g * NODES_PER + tid];
    __syncthreads();
    if (tid < NODES_PER) {
        float ki = skey[tid];
        int r = 0;
        for (int j = 0; j < NODES_PER; ++j) {
            float kj = skey[j];
            r += (kj > ki) || (kj == ki && j < tid);   // stable argsort(-key)
        }
        srank[tid] = r;
    }
    __syncthreads();
    for (int i = 0; i < NODES_PER; ++i) {
        int r = srank[i];
        if (r < KTOP) {
            const bf16* src = feat + ((size_t)g * NODES_PER + i) * LATENT;
            bf16* dst = top + ((size_t)g * KTOP + r) * LATENT;
            for (int c = tid; c < LATENT; c += blockDim.x) dst[c] = src[c];
        }
    }
}

// ------------------------------------------------- fused conv1 + maxpool + conv2 + MLP
__global__ void k_head(const bf16* __restrict__ top,
                       const float* __restrict__ cw1, const float* __restrict__ cb1,
                       const float* __restrict__ cw2, const float* __restrict__ cb2,
                       const float* __restrict__ mW1, const float* __restrict__ mb1,
                       const float* __restrict__ mW2, const float* __restrict__ mb2,
                       float* __restrict__ out) {
    __shared__ float scw1[16 * LATENT];   // 24640 B
    __shared__ float scw2[32 * 16 * 5];   // 10240 B
    __shared__ float sY1[16 * KTOP];      //  2560 B
    __shared__ float sP[16 * 20];         //  1280 B
    __shared__ float sy2[512];            //  2048 B
    __shared__ float sm1[128];            //   512 B
    int g   = blockIdx.x;
    int tid = threadIdx.x;

    for (int i = tid; i < 16 * LATENT; i += blockDim.x) scw1[i] = cw1[i];
    for (int i = tid; i < 32 * 16 * 5; i += blockDim.x) scw2[i] = cw2[i];
    __syncthreads();

    // conv1 (kernel=stride=LATENT) + relu : Y1[o,k] = relu(sum_l cw1[o,l]*top[g,k,l] + cb1[o])
    for (int idx = tid; idx < 16 * KTOP; idx += blockDim.x) {
        int k = idx >> 4, o = idx & 15;
        const bf16* row = top + ((size_t)g * KTOP + k) * LATENT;
        float acc = cb1[o];
        for (int l = 0; l < LATENT; ++l) acc += (float)row[l] * scw1[o * LATENT + l];
        sY1[o * KTOP + k] = fmaxf(acc, 0.0f);
    }
    __syncthreads();

    // maxpool 2,2 -> P[16,20]
    for (int idx = tid; idx < 16 * 20; idx += blockDim.x) {
        int i = idx / 20, t = idx % 20;
        sP[idx] = fmaxf(sY1[i * KTOP + 2 * t], sY1[i * KTOP + 2 * t + 1]);
    }
    __syncthreads();

    // conv2 (16->32 ch, kernel 5) + relu -> y2 flat [o*16+t]
    for (int idx = tid; idx < 512; idx += blockDim.x) {
        int o = idx >> 4, t = idx & 15;
        float acc = cb2[o];
#pragma unroll
        for (int i = 0; i < 16; ++i)
#pragma unroll
            for (int j = 0; j < 5; ++j)
                acc += scw2[o * 80 + i * 5 + j] * sP[i * 20 + t + j];
        sy2[idx] = fmaxf(acc, 0.0f);
    }
    __syncthreads();

    // MLP 512->128 relu
    if (tid < 128) {
        float acc = mb1[tid];
        for (int q = 0; q < 512; ++q) acc += sy2[q] * mW1[(size_t)q * 128 + tid];
        sm1[tid] = fmaxf(acc, 0.0f);
    }
    __syncthreads();

    if (tid == 0) {
        float s = mb2[0];
        for (int h = 0; h < 128; ++h) s += sm1[h] * mW2[h];
        out[g] = s;
    }
}

// ---------------------------------------------------------------- launcher
extern "C" void kernel_launch(void* const* d_in, const int* in_sizes, int n_in,
                              void* d_out, int out_size, void* d_ws, size_t ws_size,
                              hipStream_t stream) {
    const float* x       = (const float*)d_in[0];
    const int*   z       = (const int*)  d_in[1];
    const int*   ei      = (const int*)  d_in[2];
    const float* z_table = (const float*)d_in[4];
    const float* W0 = (const float*)d_in[5];  const float* b0 = (const float*)d_in[6];
    const float* W1 = (const float*)d_in[7];  const float* b1 = (const float*)d_in[8];
    const float* W2 = (const float*)d_in[9];  const float* b2 = (const float*)d_in[10];
    const float* W3 = (const float*)d_in[11]; const float* b3 = (const float*)d_in[12];
    const float* cw1 = (const float*)d_in[13]; const float* cb1 = (const float*)d_in[14];
    const float* cw2 = (const float*)d_in[15]; const float* cb2 = (const float*)d_in[16];
    const float* mW1 = (const float*)d_in[17]; const float* mb1 = (const float*)d_in[18];
    const float* mW2 = (const float*)d_in[19]; const float* mb2 = (const float*)d_in[20];
    float* out = (float*)d_out;

    char* ws = (char*)d_ws;
    size_t off = 0;
    auto alloc = [&](size_t bytes) -> char* {
        char* p = ws + off;
        off = (off + bytes + 255) & ~(size_t)255;
        return p;
    };
    int*   deg  = (int*)  alloc((size_t)NNODES * 4);
    float* dinv = (float*)alloc((size_t)NNODES * 4);
    float* norm = (float*)alloc((size_t)NEDGES * 4);
    bf16*  h0   = (bf16*) alloc((size_t)NNODES * 256 * 2);
    bf16*  hA   = (bf16*) alloc((size_t)NNODES * 128 * 2);
    bf16*  hB   = (bf16*) alloc((size_t)NNODES * 128 * 2);
    float* hW   = (float*)alloc((size_t)NNODES * 128 * 4);
    bf16*  feat = (bf16*) alloc((size_t)NNODES * LATENT * 2);
    float* keys = (float*)alloc((size_t)NNODES * 4);
    float* hw3  = (float*)alloc((size_t)NNODES * 4);
    bf16*  top  = (bf16*) alloc((size_t)NUM_GRAPHS * KTOP * LATENT * 2);
    bf16*  wt0  = (bf16*) alloc((size_t)128 * 256 * 2);
    bf16*  wt1  = (bf16*) alloc((size_t)128 * 128 * 2);
    bf16*  wt2  = (bf16*) alloc((size_t)128 * 128 * 2);
    (void)ws_size; (void)in_sizes; (void)n_in; (void)out_size;

    const int TB = 256;
    int nb_nodes = (NNODES + TB - 1) / TB;          // 391
    int nb_edges = (NEDGES + TB - 1) / TB;          // 6250

    // degree / normalization
    k_zero_i32<<<nb_nodes, TB, 0, stream>>>(deg, NNODES);
    k_deg    <<<nb_edges, TB, 0, stream>>>(ei, deg, NEDGES);
    k_dinv   <<<nb_nodes, TB, 0, stream>>>(deg, dinv, NNODES);
    k_norm   <<<nb_edges, TB, 0, stream>>>(ei, dinv, norm, NEDGES);

    // input embedding + weight transposes (f32 -> bf16)
    k_embed<<<(int)(((long)NNODES * 256 + TB - 1) / TB), TB, 0, stream>>>(z, z_table, x, h0);
    k_cvtT<<<(256 * 128 + TB - 1) / TB, TB, 0, stream>>>(W0, wt0, 256, 128);
    k_cvtT<<<(128 * 128 + TB - 1) / TB, TB, 0, stream>>>(W1, wt1, 128, 128);
    k_cvtT<<<(128 * 128 + TB - 1) / TB, TB, 0, stream>>>(W2, wt2, 128, 128);

    // GCN layers: WMMA matmul + per-graph LDS scatter (+bias,+tanh)
    int gemm_blocks = ((NNODES / 16) * (128 / 16) + 7) / 8;   // 8 waves / block
    k_gemm_wmma <<<gemm_blocks, TB, 0, stream>>>(h0, wt0, hW, NNODES, 256, 128);
    k_scatter128<<<NUM_GRAPHS, TB, 0, stream>>>(hW, ei, norm, b0, feat, 0, hA);
    k_gemm_wmma <<<gemm_blocks, TB, 0, stream>>>(hA, wt1, hW, NNODES, 128, 128);
    k_scatter128<<<NUM_GRAPHS, TB, 0, stream>>>(hW, ei, norm, b1, feat, 128, hB);
    k_gemm_wmma <<<gemm_blocks, TB, 0, stream>>>(hB, wt2, hW, NNODES, 128, 128);
    k_scatter128<<<NUM_GRAPHS, TB, 0, stream>>>(hW, ei, norm, b2, feat, 256, hA);

    // layer 3 (fout=1): matvec + scalar scatter -> feat[:,384] and sort keys
    k_matvec_w3<<<nb_nodes, TB, 0, stream>>>(hA, W3, hw3, NNODES);
    k_scatter1 <<<NUM_GRAPHS, TB, 0, stream>>>(hw3, ei, norm, b3, feat, keys);

    // global_sort_pool + fused head
    k_sortpool<<<NUM_GRAPHS, TB, 0, stream>>>(keys, feat, top);
    k_head    <<<NUM_GRAPHS, TB, 0, stream>>>(top, cw1, cb1, cw2, cb2,
                                              mW1, mb1, mW2, mb2, out);
}
